// EGNN_12610023981470
// MI455X (gfx1250) — matrix operand order repaired and checked by reading, not compiled, not run on hardware
//
#include <hip/hip_runtime.h>
#include <hip/hip_bf16.h>
#include <math.h>

// ---------------------------------------------------------------------------
// EGNN forward for MI455X (gfx1250). All dense GEMM work runs through
// v_wmma_f32_16x16x32_f16 (wave32). Edge-MLP layer-1 is decomposed into
// per-node GEMMs (linear before SiLU); layer-2 is a per-edge WMMA GEMM with
// activations staged in LDS. Edge kernels: 12 waves, wave = N-tile, two
// accumulators (M-tiles) sharing one B fragment per k-step.
// ---------------------------------------------------------------------------

typedef __attribute__((ext_vector_type(16))) _Float16 v16h;
typedef __attribute__((ext_vector_type(8)))  _Float16 v8h;
typedef __attribute__((ext_vector_type(8)))  float    v8f;

#define HDIM 192
#define NNODE 96
#define BATCH 8
#define NROW (BATCH*NNODE)          // 768
#define NEDGE (BATCH*NNODE*NNODE)   // 73728
#define NFEAT 17
#define CRANGE 3.75f                // 15.0 / 4 layers

__device__ __forceinline__ float siluf(float x)    { return x / (1.f + __expf(-x)); }
__device__ __forceinline__ float sigmoidf_(float x){ return 1.f / (1.f + __expf(-x)); }

// ---- WMMA fragment loaders (wave32; A and B both use contiguous k = 16h+s) ----------
// A 16x32 f16: lane m (0-15) rows, half=lane>>4 selects K 16h..16h+15
__device__ __forceinline__ v16h load_a_frag(const _Float16* A, int lda, int k0) {
  int lane = threadIdx.x & 31;
  int m = lane & 15, hh = lane >> 4;
  const _Float16* p = A + m * lda + k0 + 16 * hh;
  v8h lo = *(const v8h*)p;
  v8h hi = *(const v8h*)(p + 8);
  return __builtin_shufflevector(lo, hi, 0,1,2,3,4,5,6,7,8,9,10,11,12,13,14,15);
}
// B 32x16 f16 from BT stored [N][K] row-major: lane n holds K 16h..16h+15
__device__ __forceinline__ v16h load_b_frag(const _Float16* BT, int ldb, int k0) {
  int lane = threadIdx.x & 31;
  int n = lane & 15, hh = lane >> 4;
  return *(const v16h*)(BT + n * ldb + k0 + 16 * hh);
}

// ---- generic one-wave-per-tile WMMA GEMM with fused epilogue --------------
// C[m,n] = epi( A[M x K]f16 @ BT[N x K]f16 + bias ), grid=(M/16, N/16), 32 thr
__global__ void k_gemm(const _Float16* __restrict__ A, int lda,
                       const _Float16* __restrict__ BT, int ldb,
                       const float* __restrict__ bias,
                       const float* resid,            // may alias C32
                       const float* __restrict__ rowmask,
                       float* C32, _Float16* C16,
                       int ldc, int K, int act) {
  int m0 = blockIdx.x * 16, n0 = blockIdx.y * 16;
  v8f acc = {};
  const _Float16* Ab = A + m0 * lda;
  const _Float16* Bb = BT + (size_t)n0 * ldb;
  for (int k0 = 0; k0 < K; k0 += 32) {
    v16h a = load_a_frag(Ab, lda, k0);
    v16h b = load_b_frag(Bb, ldb, k0);
    acc = __builtin_amdgcn_wmma_f32_16x16x32_f16(false, a, false, b,
                                                 (short)0, acc, false, false);
  }
  int lane = threadIdx.x & 31;
  int n = n0 + (lane & 15);
  int mb = m0 + (lane >> 4) * 8;
  float bv = bias ? bias[n] : 0.f;
  #pragma unroll
  for (int r = 0; r < 8; ++r) {
    int m = mb + r;
    float v = acc[r] + bv;
    if (act) v = siluf(v);
    if (resid)   v += resid[m * ldc + n];
    if (rowmask) v *= rowmask[m];
    if (C32) C32[m * ldc + n] = v;
    if (C16) C16[m * ldc + n] = (_Float16)v;
  }
}

// ---- edge kernel for GCL ---------------------------------------------------
// One block (384 thr = 12 waves) = (b, i, 32 consecutive j).
// Phase A: t = silu(layer-1) into LDS (f16).
// Phase B: wave w = N-tile w; two accumulators (M-tiles 0/1) share B frags.
// Phase C: att = sigmoid(mij . aw + ab) * mask, 12-way split reduce.
// Phase D: agg[i] += sum_j mij*att (f32 atomics).
__global__ void k_edge_gcl(const float* __restrict__ pre_src,
                           const float* __restrict__ pre_tgt,
                           const float* __restrict__ radial,
                           const float* __restrict__ d0,
                           const float* __restrict__ ew1,   // f32 (386,192)
                           const float* __restrict__ eb1,
                           const _Float16* __restrict__ ew2t,
                           const float* __restrict__ eb2,
                           const float* __restrict__ aw,
                           const float* __restrict__ ab,
                           const float* __restrict__ emask,
                           float* __restrict__ agg) {
  __shared__ _Float16 tA[32][200];
  __shared__ float mij[32][193];
  __shared__ float part[32][13];
  __shared__ float att[32];
  int tid = threadIdx.x;
  int tile = blockIdx.x;
  int jt = tile % 3;
  int i  = (tile / 3) % NNODE;
  int b  = tile / (3 * NNODE);
  int j0 = jt * 32;
  int rowi = b * NNODE + i;
  const float* ps = pre_src + rowi * HDIM;
  for (int idx = tid; idx < 32 * HDIM; idx += 384) {
    int e = idx / HDIM, c = idx % HDIM;
    int j = j0 + e;
    int eidx = rowi * NNODE + j;
    float v = ps[c] + pre_tgt[(b * NNODE + j) * HDIM + c]
            + radial[eidx] * ew1[384 * HDIM + c]
            + d0[eidx]     * ew1[385 * HDIM + c] + eb1[c];
    tA[e][c] = (_Float16)siluf(v);
  }
  __syncthreads();
  {
    int nt = tid >> 5;                     // wave id == N tile (0..11)
    v8f acc0 = {}, acc1 = {};
    const _Float16* Bb = ew2t + nt * 16 * HDIM;
    for (int k0 = 0; k0 < HDIM; k0 += 32) {
      v16h bf = load_b_frag(Bb, HDIM, k0);
      v16h a0 = load_a_frag(&tA[0][0],  200, k0);
      v16h a1 = load_a_frag(&tA[16][0], 200, k0);
      acc0 = __builtin_amdgcn_wmma_f32_16x16x32_f16(false, a0, false, bf,
                                                    (short)0, acc0, false, false);
      acc1 = __builtin_amdgcn_wmma_f32_16x16x32_f16(false, a1, false, bf,
                                                    (short)0, acc1, false, false);
    }
    int lane = tid & 31;
    int n = nt * 16 + (lane & 15);
    int mb = (lane >> 4) * 8;
    float bv = eb2[n];
    #pragma unroll
    for (int r = 0; r < 8; ++r) {
      mij[mb + r][n]      = siluf(acc0[r] + bv);
      mij[16 + mb + r][n] = siluf(acc1[r] + bv);
    }
  }
  __syncthreads();
  {                                        // att dot: 12 segs x 16 ch per edge
    int e = tid & 31, seg = tid >> 5;
    float s = 0.f;
    #pragma unroll
    for (int c = seg * 16; c < seg * 16 + 16; ++c) s += mij[e][c] * aw[c];
    part[e][seg] = s;
  }
  __syncthreads();
  if (tid < 32) {
    float s = ab[0];
    #pragma unroll
    for (int k = 0; k < 12; ++k) s += part[tid][k];
    att[tid] = sigmoidf_(s) * emask[rowi * NNODE + j0 + tid];
  }
  __syncthreads();
  if (tid < 2 * HDIM) {
    int c = tid % HDIM, hf = tid / HDIM;
    float s = 0.f;
    #pragma unroll 4
    for (int e = hf * 16; e < hf * 16 + 16; ++e) s += mij[e][c] * att[e];
    atomicAdd(&agg[rowi * HDIM + c], s);
  }
}

// ---- edge kernel for equivariant coordinate update -------------------------
__global__ void k_edge_equiv(const float* __restrict__ pre_src,
                             const float* __restrict__ pre_tgt,
                             const float* __restrict__ radial,
                             const float* __restrict__ d0,
                             const float* __restrict__ cw1,  // f32 (386,192)
                             const float* __restrict__ cb1,
                             const _Float16* __restrict__ cw2t,
                             const float* __restrict__ cb2,
                             const float* __restrict__ cw3,
                             const float* __restrict__ emask,
                             const float* __restrict__ cdiff,
                             float* __restrict__ x_cur) {
  __shared__ _Float16 tA[32][200];
  __shared__ float phi[32][193];
  __shared__ float part[32][13];
  __shared__ float trv[32][4];
  int tid = threadIdx.x;
  int tile = blockIdx.x;
  int jt = tile % 3;
  int i  = (tile / 3) % NNODE;
  int b  = tile / (3 * NNODE);
  int j0 = jt * 32;
  int rowi = b * NNODE + i;
  const float* ps = pre_src + rowi * HDIM;
  for (int idx = tid; idx < 32 * HDIM; idx += 384) {
    int e = idx / HDIM, c = idx % HDIM;
    int j = j0 + e;
    int eidx = rowi * NNODE + j;
    float v = ps[c] + pre_tgt[(b * NNODE + j) * HDIM + c]
            + radial[eidx] * cw1[384 * HDIM + c]
            + d0[eidx]     * cw1[385 * HDIM + c] + cb1[c];
    tA[e][c] = (_Float16)siluf(v);
  }
  __syncthreads();
  {
    int nt = tid >> 5;
    v8f acc0 = {}, acc1 = {};
    const _Float16* Bb = cw2t + nt * 16 * HDIM;
    for (int k0 = 0; k0 < HDIM; k0 += 32) {
      v16h bf = load_b_frag(Bb, HDIM, k0);
      v16h a0 = load_a_frag(&tA[0][0],  200, k0);
      v16h a1 = load_a_frag(&tA[16][0], 200, k0);
      acc0 = __builtin_amdgcn_wmma_f32_16x16x32_f16(false, a0, false, bf,
                                                    (short)0, acc0, false, false);
      acc1 = __builtin_amdgcn_wmma_f32_16x16x32_f16(false, a1, false, bf,
                                                    (short)0, acc1, false, false);
    }
    int lane = tid & 31;
    int n = nt * 16 + (lane & 15);
    int mb = (lane >> 4) * 8;
    float bv = cb2[n];
    #pragma unroll
    for (int r = 0; r < 8; ++r) {
      phi[mb + r][n]      = siluf(acc0[r] + bv);
      phi[16 + mb + r][n] = siluf(acc1[r] + bv);
    }
  }
  __syncthreads();
  {
    int e = tid & 31, seg = tid >> 5;
    float s = 0.f;
    #pragma unroll
    for (int c = seg * 16; c < seg * 16 + 16; ++c) s += phi[e][c] * cw3[c];
    part[e][seg] = s;
  }
  __syncthreads();
  if (tid < 32) {
    float s = 0.f;
    #pragma unroll
    for (int k = 0; k < 12; ++k) s += part[tid][k];
    int j = j0 + tid;
    int eidx = rowi * NNODE + j;
    float tr = tanhf(s) * CRANGE * emask[eidx];
    for (int d = 0; d < 3; ++d) trv[tid][d] = cdiff[eidx * 3 + d] * tr;
  }
  __syncthreads();
  if (tid < 3) {
    float s = 0.f;
    for (int e = 0; e < 32; ++e) s += trv[e][tid];
    atomicAdd(&x_cur[rowi * 3 + tid], s);
  }
}

// ---- small scalar kernels --------------------------------------------------
__global__ void k_transpose(const float* __restrict__ W, _Float16* __restrict__ Wt,
                            int K, int N) {    // W:[K][N] -> Wt:[N][K] f16
  int idx = blockIdx.x * 256 + threadIdx.x;
  if (idx >= K * N) return;
  int n = idx / K, k = idx % K;
  Wt[n * K + k] = (_Float16)W[k * N + n];
}

__global__ void k_zero(float* p, int n) {
  int idx = blockIdx.x * 256 + threadIdx.x;
  if (idx < n) p[idx] = 0.f;
}

__global__ void k_copy(const float* a, float* b, int n) {
  int idx = blockIdx.x * 256 + threadIdx.x;
  if (idx < n) b[idx] = a[idx];
}

__global__ void k_coord(const float* __restrict__ x, float* __restrict__ radial,
                        float* __restrict__ cdiff) {
  int idx = blockIdx.x * 256 + threadIdx.x;
  if (idx >= NEDGE) return;
  int b = idx / (NNODE * NNODE);
  int r = idx % (NNODE * NNODE);
  int i = r / NNODE, j = r % NNODE;
  float dx[3]; float r2 = 0.f;
  #pragma unroll
  for (int d = 0; d < 3; ++d) {
    dx[d] = x[(b * NNODE + i) * 3 + d] - x[(b * NNODE + j) * 3 + d];
    r2 += dx[d] * dx[d];
  }
  radial[idx] = r2;
  float inv = 1.f / (sqrtf(r2 + 1e-8f) + 1.f);
  #pragma unroll
  for (int d = 0; d < 3; ++d) cdiff[idx * 3 + d] = dx[d] * inv;
}

__global__ void k_embed(const float* __restrict__ hin, const float* __restrict__ Win,
                        const float* __restrict__ bin, const float* __restrict__ flags,
                        float* __restrict__ h32, _Float16* __restrict__ h16) {
  int idx = blockIdx.x * 256 + threadIdx.x;
  if (idx >= NROW * HDIM) return;
  int row = idx / HDIM, c = idx % HDIM;
  float s = bin[c];
  for (int f = 0; f < NFEAT; ++f) s += hin[row * NFEAT + f] * Win[f * HDIM + c];
  s *= flags[row];
  h32[idx] = s; h16[idx] = (_Float16)s;
}

__global__ void k_hcat(const float* __restrict__ h32, const float* __restrict__ agg,
                       _Float16* __restrict__ hcat) {
  int idx = blockIdx.x * 256 + threadIdx.x;
  if (idx >= NROW * 2 * HDIM) return;
  int row = idx / (2 * HDIM), c = idx % (2 * HDIM);
  float v = (c < HDIM) ? h32[row * HDIM + c] : agg[row * HDIM + c - HDIM];
  hcat[idx] = (_Float16)v;
}

__global__ void k_final(const float* __restrict__ h32, const float* __restrict__ Wout,
                        const float* __restrict__ bout, const float* __restrict__ flags,
                        float* __restrict__ out) {
  int row = blockIdx.x * 64 + threadIdx.x;
  if (row >= NROW) return;
  float hv[NFEAT];
  for (int o = 0; o < NFEAT; ++o) {
    float s = bout[o];
    for (int k = 0; k < HDIM; ++k) s += h32[row * HDIM + k] * Wout[k * NFEAT + o];
    hv[o] = s * flags[row];
  }
  float mx = hv[0];
  for (int o = 1; o < 16; ++o) mx = fmaxf(mx, hv[o]);
  float e[16], sum = 0.f;
  for (int o = 0; o < 16; ++o) { e[o] = __expf(hv[o] - mx); sum += e[o]; }
  float inv = 1.f / sum;
  for (int o = 0; o < 16; ++o) out[row * NFEAT + o] = e[o] * inv * flags[row];
  out[row * NFEAT + 16] = hv[16];
}

__global__ void k_xout(const float* __restrict__ xc, const float* __restrict__ x0,
                       float* __restrict__ out) {
  int idx = blockIdx.x * 256 + threadIdx.x;
  if (idx < NROW * 3) out[idx] = xc[idx] - x0[idx];
}

// ---------------------------------------------------------------------------
extern "C" void kernel_launch(void* const* d_in, const int* in_sizes, int n_in,
                              void* d_out, int out_size, void* d_ws, size_t ws_size,
                              hipStream_t stream) {
  (void)in_sizes; (void)n_in; (void)out_size; (void)ws_size;
  const float* hin   = (const float*)d_in[0];
  const float* xin   = (const float*)d_in[1];
  const float* flags = (const float*)d_in[2];
  const float* emask = (const float*)d_in[3];
  // params flattened as a JAX pytree: dict keys sorted. Top: Win, Wout, bin,
  // bout, layers[0..3]; layer dict: equiv < gcl0 < gcl1; equiv: cb1,cb2,cw1,
  // cw2,cw3; gcl: ab,aw,eb1,eb2,ew1,ew2,nb1,nb2,nw1,nw2.
  int p = 4;
  const float* Win  = (const float*)d_in[p++];
  const float* Wout = (const float*)d_in[p++];
  const float* bin  = (const float*)d_in[p++];
  const float* bout = (const float*)d_in[p++];
  struct EQ { const float *cb1,*cb2,*cw1,*cw2,*cw3; } eq[4];
  struct GC { const float *ab,*aw,*eb1,*eb2,*ew1,*ew2,*nb1,*nb2,*nw1,*nw2; } gc[4][2];
  for (int l = 0; l < 4; ++l) {
    eq[l].cb1 = (const float*)d_in[p++]; eq[l].cb2 = (const float*)d_in[p++];
    eq[l].cw1 = (const float*)d_in[p++]; eq[l].cw2 = (const float*)d_in[p++];
    eq[l].cw3 = (const float*)d_in[p++];
    for (int g = 0; g < 2; ++g) {
      gc[l][g].ab  = (const float*)d_in[p++]; gc[l][g].aw  = (const float*)d_in[p++];
      gc[l][g].eb1 = (const float*)d_in[p++]; gc[l][g].eb2 = (const float*)d_in[p++];
      gc[l][g].ew1 = (const float*)d_in[p++]; gc[l][g].ew2 = (const float*)d_in[p++];
      gc[l][g].nb1 = (const float*)d_in[p++]; gc[l][g].nb2 = (const float*)d_in[p++];
      gc[l][g].nw1 = (const float*)d_in[p++]; gc[l][g].nw2 = (const float*)d_in[p++];
    }
  }

  // bump allocator over d_ws (256B aligned chunks)
  char* base = (char*)d_ws; size_t off = 0;
  auto alloc = [&](size_t bytes) -> void* {
    off = (off + 255) & ~(size_t)255;
    void* r = base + off; off += bytes; return r;
  };
  float*    h32    = (float*)   alloc((size_t)NROW * HDIM * 4);
  _Float16* h16    = (_Float16*)alloc((size_t)NROW * HDIM * 2);
  float*    pre_s  = (float*)   alloc((size_t)NROW * HDIM * 4);
  float*    pre_t  = (float*)   alloc((size_t)NROW * HDIM * 4);
  float*    agg    = (float*)   alloc((size_t)NROW * HDIM * 4);
  _Float16* hcat16 = (_Float16*)alloc((size_t)NROW * 2 * HDIM * 2);
  _Float16* u16    = (_Float16*)alloc((size_t)NROW * HDIM * 2);
  float*    radial = (float*)   alloc((size_t)NEDGE * 4);
  float*    d0     = (float*)   alloc((size_t)NEDGE * 4);
  float*    cdiff  = (float*)   alloc((size_t)NEDGE * 3 * 4);
  float*    x_cur  = (float*)   alloc((size_t)NROW * 3 * 4);
  const size_t WU = (size_t)HDIM * HDIM;   // 36864 halfs
  _Float16 *ew1sT[4][2], *ew1tT[4][2], *ew2T[4][2], *nw1T[4][2], *nw2T[4][2];
  _Float16 *cw1sT[4], *cw1tT[4], *cw2T[4];
  for (int l = 0; l < 4; ++l) {
    for (int g = 0; g < 2; ++g) {
      ew1sT[l][g] = (_Float16*)alloc(WU * 2);
      ew1tT[l][g] = (_Float16*)alloc(WU * 2);
      ew2T [l][g] = (_Float16*)alloc(WU * 2);
      nw1T [l][g] = (_Float16*)alloc(WU * 4);   // [192][384]
      nw2T [l][g] = (_Float16*)alloc(WU * 2);
    }
    cw1sT[l] = (_Float16*)alloc(WU * 2);
    cw1tT[l] = (_Float16*)alloc(WU * 2);
    cw2T [l] = (_Float16*)alloc(WU * 2);
  }

  auto T = [&](const float* W, _Float16* Wt, int K, int N) {
    int tot = K * N;
    k_transpose<<<dim3((tot + 255) / 256), dim3(256), 0, stream>>>(W, Wt, K, N);
  };
  // weight prep (deterministic, re-done every call)
  for (int l = 0; l < 4; ++l) {
    for (int g = 0; g < 2; ++g) {
      T(gc[l][g].ew1,              ew1sT[l][g], HDIM, HDIM);
      T(gc[l][g].ew1 + HDIM*HDIM,  ew1tT[l][g], HDIM, HDIM);
      T(gc[l][g].ew2,              ew2T [l][g], HDIM, HDIM);
      T(gc[l][g].nw1,              nw1T [l][g], 2*HDIM, HDIM);
      T(gc[l][g].nw2,              nw2T [l][g], HDIM, HDIM);
    }
    T(eq[l].cw1,             cw1sT[l], HDIM, HDIM);
    T(eq[l].cw1 + HDIM*HDIM, cw1tT[l], HDIM, HDIM);
    T(eq[l].cw2,             cw2T [l], HDIM, HDIM);
  }

  k_embed<<<dim3((NROW*HDIM + 255)/256), dim3(256), 0, stream>>>(hin, Win, bin, flags, h32, h16);
  k_copy <<<dim3((NROW*3 + 255)/256),    dim3(256), 0, stream>>>(xin, x_cur, NROW*3);
  k_coord<<<dim3((NEDGE + 255)/256),     dim3(256), 0, stream>>>(x_cur, d0, cdiff);

  dim3 gN(NROW/16, HDIM/16), b1(32), bE(384), gE(BATCH*NNODE*3);
  for (int l = 0; l < 4; ++l) {
    k_coord<<<dim3((NEDGE + 255)/256), dim3(256), 0, stream>>>(x_cur, radial, cdiff);
    for (int g = 0; g < 2; ++g) {
      const GC& W = gc[l][g];
      k_gemm<<<gN, b1, 0, stream>>>(h16, HDIM, ew1sT[l][g], HDIM, nullptr, nullptr, nullptr,
                                    pre_s, nullptr, HDIM, HDIM, 0);
      k_gemm<<<gN, b1, 0, stream>>>(h16, HDIM, ew1tT[l][g], HDIM, nullptr, nullptr, nullptr,
                                    pre_t, nullptr, HDIM, HDIM, 0);
      k_zero<<<dim3((NROW*HDIM + 255)/256), dim3(256), 0, stream>>>(agg, NROW*HDIM);
      k_edge_gcl<<<gE, bE, 0, stream>>>(
          pre_s, pre_t, radial, d0, W.ew1, W.eb1, ew2T[l][g], W.eb2, W.aw, W.ab, emask, agg);
      k_hcat<<<dim3((NROW*2*HDIM + 255)/256), dim3(256), 0, stream>>>(h32, agg, hcat16);
      k_gemm<<<gN, b1, 0, stream>>>(hcat16, 2*HDIM, nw1T[l][g], 2*HDIM, W.nb1, nullptr, nullptr,
                                    nullptr, u16, HDIM, 2*HDIM, 1);
      k_gemm<<<gN, b1, 0, stream>>>(u16, HDIM, nw2T[l][g], HDIM, W.nb2, h32, flags,
                                    h32, h16, HDIM, HDIM, 0);
    }
    k_gemm<<<gN, b1, 0, stream>>>(h16, HDIM, cw1sT[l], HDIM, nullptr, nullptr, nullptr,
                                  pre_s, nullptr, HDIM, HDIM, 0);
    k_gemm<<<gN, b1, 0, stream>>>(h16, HDIM, cw1tT[l], HDIM, nullptr, nullptr, nullptr,
                                  pre_t, nullptr, HDIM, HDIM, 0);
    k_edge_equiv<<<gE, bE, 0, stream>>>(
        pre_s, pre_t, radial, d0, eq[l].cw1, eq[l].cb1, cw2T[l], eq[l].cb2, eq[l].cw3,
        emask, cdiff, x_cur);
  }

  float* out = (float*)d_out;
  k_final<<<dim3((NROW + 63)/64), dim3(64), 0, stream>>>(h32, Wout, bout, flags, out);
  k_xout <<<dim3((NROW*3 + 255)/256), dim3(256), 0, stream>>>(x_cur, xin, out + NROW*NFEAT);
}
